// DynamicGraph_88776974008834
// MI455X (gfx1250) — compile-verified, matrix operand order checked by reading, not run on hardware
//
#include <hip/hip_runtime.h>
#include <hip/hip_bf16.h>

// ---------------------------------------------------------------------------
// CDNA5 (gfx1250) implementation of the DynamicGraph forward pass.
// All large GEMMs run on v_wmma_f32_16x16x32_bf16 (wave32, 16x64 strip/wave).
// ---------------------------------------------------------------------------

typedef __attribute__((ext_vector_type(16))) __bf16 v16bf;
typedef __attribute__((ext_vector_type(8)))  float  v8f;

static constexpr int BB   = 16;     // B_
static constexpr int TT   = 8;      // T_
static constexpr int HW   = 196;    // 14*14
static constexpr int HH   = 14;
static constexpr int NN   = 9;      // nodes
static constexpr int CBC  = 1024;   // backbone dim
static constexpr int CC   = 512;    // node dim
static constexpr int DD   = 128;    // lstm dim
static constexpr int MPX  = 128 * HW;   // 25088 pixel rows
static constexpr int MBN  = 128 * NN;   // 1152 (bt,n) rows
static constexpr int MRN  = BB * NN;    // 144 recurrent rows

// ---------------------------------------------------------------------------
// WMMA GEMM:  C[M,N] = A[M,Kp] * B[N,Kp]^T
//   A row-major bf16 [M, lda],  M % 16 == 0 (guaranteed by caller)
//   B row-major bf16 [Nalloc, ldb], rows zero-padded to a multiple of 64
//   Kp % 32 == 0 (K zero-padded at conversion time)
// Each wave computes a 16x64 strip: A fragment reused across 4 wmmas/k-step.
// epilogue modes:
//   0: C = acc + bias
//   1: C = acc + bias + resid            (resid same [M,ldc] layout)
//   2: C = resid + relu(acc + bias)
//   3: scatter channel-major: out[(m/196*1024 + n)*196 + m%196] = resid[..] + acc + bias[n]
// ---------------------------------------------------------------------------
struct FragBF { union { v16bf v; uint4 q[2]; }; };

__device__ __forceinline__ FragBF load_frag_bf(const __bf16* __restrict__ base,
                                               int row, int ld, int kbase) {
  FragBF f;
  const uint4* p = reinterpret_cast<const uint4*>(base + (size_t)row * ld + kbase);
  f.q[0] = p[0];   // K +0..7   (16B)
  f.q[1] = p[2];   // K +16..23 (16B)
  return f;
}

__global__ __launch_bounds__(256) void k_wmma_gemm(
    const __bf16* __restrict__ A, const __bf16* __restrict__ Bw,
    float* __restrict__ C,
    const float* __restrict__ bias, const float* __restrict__ resid,
    int M, int N, int Kp, int lda, int ldb, int ldc, int mode) {
  int waveId = (int)((blockIdx.x * blockDim.x + threadIdx.x) >> 5);
  int lane   = threadIdx.x & 31;
  int strips = (N + 63) >> 6;          // 64-wide N strips
  int mt = waveId / strips;
  int nt = waveId % strips;
  if (mt * 16 >= M) return;

  int half = lane >> 4;        // 0: K 0..7/16..23 group, 1: K 8..15/24..31 group
  int lr   = lane & 15;
  int arow = mt * 16 + lr;
  int brow = nt * 64 + lr;

  v8f acc0 = {0.f, 0.f, 0.f, 0.f, 0.f, 0.f, 0.f, 0.f};
  v8f acc1 = acc0, acc2 = acc0, acc3 = acc0;

  for (int kk = 0; kk < Kp; kk += 32) {
    int kb = kk + half * 8;
    FragBF fa  = load_frag_bf(A,  arow,      lda, kb);
    FragBF fb0 = load_frag_bf(Bw, brow,      ldb, kb);
    FragBF fb1 = load_frag_bf(Bw, brow + 16, ldb, kb);
    FragBF fb2 = load_frag_bf(Bw, brow + 32, ldb, kb);
    FragBF fb3 = load_frag_bf(Bw, brow + 48, ldb, kb);
    acc0 = __builtin_amdgcn_wmma_f32_16x16x32_bf16(false, fa.v, false, fb0.v, (short)0, acc0, false, false);
    acc1 = __builtin_amdgcn_wmma_f32_16x16x32_bf16(false, fa.v, false, fb1.v, (short)0, acc1, false, false);
    acc2 = __builtin_amdgcn_wmma_f32_16x16x32_bf16(false, fa.v, false, fb2.v, (short)0, acc2, false, false);
    acc3 = __builtin_amdgcn_wmma_f32_16x16x32_bf16(false, fa.v, false, fb3.v, (short)0, acc3, false, false);
  }

  for (int s = 0; s < 4; ++s) {
    v8f acc = (s == 0) ? acc0 : (s == 1) ? acc1 : (s == 2) ? acc2 : acc3;
    int col = nt * 64 + s * 16 + lr;
    if (col >= N) continue;
    float bval = (bias != nullptr) ? bias[col] : 0.f;
    for (int r = 0; r < 8; ++r) {
      int row = mt * 16 + r + half * 8;
      float val = acc[r] + bval;
      if (mode == 0) {
        C[(size_t)row * ldc + col] = val;
      } else if (mode == 1) {
        C[(size_t)row * ldc + col] = val + resid[(size_t)row * ldc + col];
      } else if (mode == 2) {
        C[(size_t)row * ldc + col] = resid[(size_t)row * ldc + col] + fmaxf(val, 0.f);
      } else { // mode 3: channel-major scatter with residual (final projection)
        int b = row / HW, p = row % HW;
        size_t idx = ((size_t)b * CBC + col) * HW + p;
        C[idx] = resid[idx] + val;
      }
    }
  }
}

// ---------------------------------------------------------------------------
// small helper kernels
// ---------------------------------------------------------------------------
// rows < rows_valid copy from src (cols < cols_valid, else 0); rows >= rows_valid -> 0
__global__ void k_tobf16(const float* __restrict__ src, __bf16* __restrict__ dst,
                         int rows_valid, int cols, int sld, int dld, int rows_total) {
  size_t idx = (size_t)blockIdx.x * blockDim.x + threadIdx.x;
  size_t tot = (size_t)rows_total * dld;
  if (idx >= tot) return;
  int r = (int)(idx / dld), c = (int)(idx % dld);
  float v = (r < rows_valid && c < cols) ? src[(size_t)r * sld + c] : 0.f;
  dst[idx] = (__bf16)v;
}

// x [BT, CB, H, W] -> pixel-major bf16 [BT*HW, CB]
__global__ void k_x_to_pxbf(const float* __restrict__ x, __bf16* __restrict__ dst) {
  size_t idx = (size_t)blockIdx.x * blockDim.x + threadIdx.x;
  size_t tot = (size_t)MPX * CBC;
  if (idx >= tot) return;
  int c  = (int)(idx % CBC);
  int bp = (int)(idx / CBC);
  int b  = bp / HW, p = bp % HW;
  dst[idx] = (__bf16)x[((size_t)b * CBC + c) * HW + p];
}

__global__ void k_fill0(float* __restrict__ p, size_t n) {
  size_t idx = (size_t)blockIdx.x * blockDim.x + threadIdx.x;
  if (idx < n) p[idx] = 0.f;
}

// softmax over the 196 spatial positions for each (bt, n); logits [(bt*196+p), 9]
__global__ void k_softmax_p(const float* __restrict__ logits, float* __restrict__ attw) {
  int bn = blockIdx.x;                 // 0..1151
  int b = bn / NN, n = bn % NN;
  __shared__ float red[256];
  int tid = threadIdx.x;
  float m = -1e30f;
  for (int p = tid; p < HW; p += 256)
    m = fmaxf(m, logits[((size_t)b * HW + p) * NN + n]);
  red[tid] = m; __syncthreads();
  for (int s = 128; s > 0; s >>= 1) { if (tid < s) red[tid] = fmaxf(red[tid], red[tid + s]); __syncthreads(); }
  m = red[0]; __syncthreads();
  float sum = 0.f;
  for (int p = tid; p < HW; p += 256) {
    float e = __expf(logits[((size_t)b * HW + p) * NN + n] - m);
    attw[(size_t)bn * HW + p] = e;
    sum += e;
  }
  red[tid] = sum; __syncthreads();
  for (int s = 128; s > 0; s >>= 1) { if (tid < s) red[tid] += red[tid + s]; __syncthreads(); }
  float inv = 1.f / red[0];
  __syncthreads();
  for (int p = tid; p < HW; p += 256) attw[(size_t)bn * HW + p] *= inv;
}

// out[bn, c] = scale * sum_p w[bn, p] * feat[(b*196+p), c]      (C=512, block=256)
__global__ void k_wpool(const float* __restrict__ w, const float* __restrict__ feat,
                        float* __restrict__ out, float scale) {
  int bn = blockIdx.x;
  int b  = bn / NN;
  int c  = threadIdx.x;
  float a0 = 0.f, a1 = 0.f;
  for (int p = 0; p < HW; ++p) {
    float wv = w[(size_t)bn * HW + p];
    const float* fp = feat + ((size_t)b * HW + p) * CC;
    a0 += wv * fp[c];
    a1 += wv * fp[c + 256];
  }
  out[(size_t)bn * CC + c]       = a0 * scale;
  out[(size_t)bn * CC + c + 256] = a1 * scale;
}

// layernorm over D=128 with per-node gain/bias (block = 128 threads, one row)
__global__ void k_layernorm(const float* __restrict__ in, float* __restrict__ out,
                            const float* __restrict__ g, const float* __restrict__ beta) {
  int row = blockIdx.x;
  int n   = row % NN;
  int tid = threadIdx.x;
  __shared__ float red[128];
  float x = in[(size_t)row * DD + tid];
  red[tid] = x; __syncthreads();
  for (int s = 64; s > 0; s >>= 1) { if (tid < s) red[tid] += red[tid + s]; __syncthreads(); }
  float mean = red[0] / DD; __syncthreads();
  float d = x - mean;
  red[tid] = d * d; __syncthreads();
  for (int s = 64; s > 0; s >>= 1) { if (tid < s) red[tid] += red[tid + s]; __syncthreads(); }
  float var = red[0] / DD;
  out[(size_t)row * DD + tid] = d * rsqrtf(var + 1e-5f) * g[n * DD + tid] + beta[n * DD + tid];
}

// per (bt,n): offsets -> anchor kernels khs[14], kws[14], kern2d[196]
__global__ void k_anchors(const float* __restrict__ ysln,
                          const float* __restrict__ Wpred, const float* __restrict__ bpred,
                          const float* __restrict__ alpha,
                          float* __restrict__ khs, float* __restrict__ kws,
                          float* __restrict__ kern2d) {
  int bn = blockIdx.x;
  int n  = bn % NN;
  __shared__ float off[4];
  __shared__ float skh[HH], skw[HH];
  int tid = threadIdx.x;
  if (tid < 4) {
    float acc = bpred[tid];
    const float* r = ysln + (size_t)bn * DD;
    const float* wr = Wpred + tid * DD;
    for (int d = 0; d < DD; ++d) acc += wr[d] * r[d];
    off[tid] = acc * alpha[tid];
  }
  __syncthreads();
  if (tid < 2 * HH) {
    int isW = tid / HH, G = tid % HH;
    const float cell = 14.f / 3.f;
    float ch, s;
    if (!isW) { ch = ((n / 3) + 0.5f) * cell + off[0]; s = cell * __expf(off[2]); }
    else      { ch = ((n % 3) + 0.5f) * cell + off[1]; s = cell * __expf(off[3]); }
    float acc = 0.f;
    for (int i = 0; i < 3; ++i) {
      float rel = (i + 0.5f) / 3.f - 0.5f;
      float pos = ch + s * rel;
      acc += fmaxf(0.f, 1.f - fabsf(pos - (float)G));
    }
    if (!isW) { skh[G] = acc; khs[(size_t)bn * HH + G] = acc; }
    else      { skw[G] = acc; kws[(size_t)bn * HH + G] = acc; }
  }
  __syncthreads();
  if (tid < HW) {
    int h = tid / HH, l = tid % HH;
    kern2d[(size_t)bn * HW + tid] = skh[h] * skw[l];
  }
}

__device__ __forceinline__ float sigf(float x) { return 1.f / (1.f + __expf(-x)); }

// LSTM elementwise step: gx pre-biased with b_ih, gh pre-biased with b_hh
__global__ void k_lstm_step(const float* __restrict__ gx, const float* __restrict__ gh,
                            float* __restrict__ h, float* __restrict__ c,
                            float* __restrict__ ys, int t) {
  int idx = blockIdx.x * blockDim.x + threadIdx.x;
  if (idx >= MRN * DD) return;
  int r = idx / DD, d = idx % DD;
  int b = r / NN, n = r % NN;
  size_t gxr = ((size_t)(b * TT + t) * NN + n) * (4 * DD);
  size_t ghr = (size_t)r * (4 * DD);
  float gi = gx[gxr + d]            + gh[ghr + d];
  float gf = gx[gxr + DD + d]       + gh[ghr + DD + d];
  float gg = gx[gxr + 2 * DD + d]   + gh[ghr + 2 * DD + d];
  float go = gx[gxr + 3 * DD + d]   + gh[ghr + 3 * DD + d];
  float cn = sigf(gf) * c[idx] + sigf(gi) * tanhf(gg);
  float hn = sigf(go) * tanhf(cn);
  c[idx] = cn; h[idx] = hn;
  ys[((size_t)(b * TT + t) * NN + n) * DD + d] = hn;
}

// GRU elementwise step: gi pre-biased with b_gih, gh pre-biased with b_ghh
__global__ void k_gru_step(const float* __restrict__ gi, const float* __restrict__ gh,
                           float* __restrict__ h, float* __restrict__ outn, int t) {
  int idx = blockIdx.x * blockDim.x + threadIdx.x;
  if (idx >= MRN * CC) return;
  int r = idx / CC, cidx = idx % CC;
  int b = r / NN, n = r % NN;
  size_t gir = ((size_t)(b * TT + t) * NN + n) * (3 * CC);
  size_t ghr = (size_t)r * (3 * CC);
  float ir = gi[gir + cidx],          hr = gh[ghr + cidx];
  float iz = gi[gir + CC + cidx],     hz = gh[ghr + CC + cidx];
  float ig = gi[gir + 2 * CC + cidx], hg = gh[ghr + 2 * CC + cidx];
  float rr = sigf(ir + hr);
  float zz = sigf(iz + hz);
  float ng = tanhf(ig + rr * hg);
  float hp = h[idx];
  float hn = (1.f - zz) * ng + zz * hp;
  h[idx] = hn;
  outn[((size_t)(b * TT + t) * NN + n) * CC + cidx] = hn;
}

// 9-node full attention per (b,t): block = 256 threads
__global__ void k_attn(const float* __restrict__ q, const float* __restrict__ k,
                       const float* __restrict__ v, float* __restrict__ agg) {
  int bt = blockIdx.x;
  __shared__ float s[NN * NN];
  int tid = threadIdx.x;
  if (tid < NN * NN) {
    int i = tid / NN, j = tid % NN;
    const float* qi = q + ((size_t)bt * NN + i) * CC;
    const float* kj = k + ((size_t)bt * NN + j) * CC;
    float acc = 0.f;
    for (int c = 0; c < CC; ++c) acc += qi[c] * kj[c];
    s[tid] = acc * 0.044194173824159216f; // 1/sqrt(512)
  }
  __syncthreads();
  if (tid < NN) {
    float m = -1e30f;
    for (int j = 0; j < NN; ++j) m = fmaxf(m, s[tid * NN + j]);
    float e[NN]; float sum = 0.f;
    for (int j = 0; j < NN; ++j) { e[j] = __expf(s[tid * NN + j] - m); sum += e[j]; }
    float inv = 1.f / sum;
    for (int j = 0; j < NN; ++j) s[tid * NN + j] = e[j] * inv;
  }
  __syncthreads();
  for (int idx = tid; idx < NN * CC; idx += 256) {
    int i = idx / CC, c = idx % CC;
    float acc = 0.f;
    for (int j = 0; j < NN; ++j)
      acc += s[i * NN + j] * v[((size_t)bt * NN + j) * CC + c];
    agg[((size_t)bt * NN + i) * CC + c] = acc;
  }
}

// remap node states to spatial map: outmap[(bt*196+p), c] = num/den  (block=256)
__global__ void k_remap(const float* __restrict__ khs, const float* __restrict__ kws,
                        const float* __restrict__ nodes, float* __restrict__ outmap) {
  int bp = blockIdx.x;                 // 0..25087
  int b = bp / HW, p = bp % HW;
  int h = p / HH, l = p % HH;
  __shared__ float wn[NN];
  __shared__ float inv_den;
  int tid = threadIdx.x;
  if (tid < NN)
    wn[tid] = khs[((size_t)b * NN + tid) * HH + h] * kws[((size_t)b * NN + tid) * HH + l];
  __syncthreads();
  if (tid == 0) {
    float d = 1e-6f;
    for (int n = 0; n < NN; ++n) d += wn[n];
    inv_den = 1.f / d;
  }
  __syncthreads();
  int c = tid;
  float a0 = 0.f, a1 = 0.f;
  for (int n = 0; n < NN; ++n) {
    const float* nr = nodes + ((size_t)b * NN + n) * CC;
    a0 += wn[n] * nr[c];
    a1 += wn[n] * nr[c + 256];
  }
  outmap[(size_t)bp * CC + c]       = a0 * inv_den;
  outmap[(size_t)bp * CC + c + 256] = a1 * inv_den;
}

// ---------------------------------------------------------------------------
extern "C" void kernel_launch(void* const* d_in, const int* in_sizes, int n_in,
                              void* d_out, int out_size, void* d_ws, size_t ws_size,
                              hipStream_t stream) {
  const float* x      = (const float*)d_in[0];
  const float* W_proj = (const float*)d_in[1];
  const float* b_proj = (const float*)d_in[2];
  const float* W_attn = (const float*)d_in[3];
  const float* W_gf   = (const float*)d_in[4];
  const float* b_gf   = (const float*)d_in[5];
  const float* g1     = (const float*)d_in[6];
  const float* beta1  = (const float*)d_in[7];
  const float* g2     = (const float*)d_in[8];
  const float* beta2  = (const float*)d_in[9];
  const float* W_ih   = (const float*)d_in[10];
  const float* W_hh   = (const float*)d_in[11];
  const float* b_ih   = (const float*)d_in[12];
  const float* b_hh   = (const float*)d_in[13];
  const float* W_pred = (const float*)d_in[14];
  const float* b_pred = (const float*)d_in[15];
  const float* alpha  = (const float*)d_in[16];
  const float* W_kp1  = (const float*)d_in[17];
  const float* b_kp1  = (const float*)d_in[18];
  const float* Wq     = (const float*)d_in[19];
  const float* Wk     = (const float*)d_in[20];
  const float* Wv     = (const float*)d_in[21];
  const float* Wu     = (const float*)d_in[22];
  const float* b_u    = (const float*)d_in[23];
  const float* W_gih  = (const float*)d_in[24];
  const float* W_ghh  = (const float*)d_in[25];
  const float* b_gih  = (const float*)d_in[26];
  const float* b_ghh  = (const float*)d_in[27];
  const float* W_back = (const float*)d_in[28];
  const float* b_back = (const float*)d_in[29];
  float* out = (float*)d_out;

  // ---- workspace arena ----------------------------------------------------
  char* base = (char*)d_ws;
  size_t off = 0;
  auto alloc = [&](size_t bytes) -> void* {
    void* p = base + off;
    off = (off + bytes + 255) & ~(size_t)255;
    return p;
  };

  float*  feat_px   = (float*)alloc((size_t)MPX * CC * 4);          // proj output, pixel-major
  void*   big1      = alloc((size_t)MPX * CBC * 2);                 // x_bf16 / out_map alias
  __bf16* x_bf      = (__bf16*)big1;
  float*  outmap    = (float*)big1;                                 // MPX*CC*4 == MPX*CBC*2
  void*   big2      = alloc((size_t)MPX * CC * 2);                  // feat_bf / outmap_bf alias
  __bf16* feat_bf   = (__bf16*)big2;
  __bf16* outmap_bf = (__bf16*)big2;
  float*  logits    = (float*)alloc((size_t)MPX * NN * 4);
  float*  attw      = (float*)alloc((size_t)MBN * HW * 4);
  float*  gf        = (float*)alloc((size_t)MBN * CC * 4);
  __bf16* gf_bf     = (__bf16*)alloc((size_t)MBN * CC * 2);
  float*  gfd       = (float*)alloc((size_t)MBN * DD * 4);
  float*  gfd_ln    = (float*)alloc((size_t)MBN * DD * 4);
  __bf16* gfdln_bf  = (__bf16*)alloc((size_t)MBN * DD * 2);
  float*  gx        = (float*)alloc((size_t)MBN * 4 * DD * 4);
  float*  hL        = (float*)alloc((size_t)MRN * DD * 4);
  float*  cL        = (float*)alloc((size_t)MRN * DD * 4);
  __bf16* hL_bf     = (__bf16*)alloc((size_t)MRN * DD * 2);
  float*  ghL       = (float*)alloc((size_t)MRN * 4 * DD * 4);
  float*  ys        = (float*)alloc((size_t)MBN * DD * 4);
  float*  ys_ln     = (float*)alloc((size_t)MBN * DD * 4);
  float*  khs       = (float*)alloc((size_t)MBN * HH * 4);
  float*  kws       = (float*)alloc((size_t)MBN * HH * 4);
  float*  kern2d    = (float*)alloc((size_t)MBN * HW * 4);
  __bf16* kern2d_bf = (__bf16*)alloc((size_t)MBN * 224 * 2);
  float*  samp      = (float*)alloc((size_t)MBN * CC * 4);
  float*  nodes     = (float*)alloc((size_t)MBN * CC * 4);
  __bf16* nodes_bf  = (__bf16*)alloc((size_t)MBN * CC * 2);
  float*  qb        = (float*)alloc((size_t)MBN * CC * 4);
  float*  kb        = (float*)alloc((size_t)MBN * CC * 4);
  float*  vb        = (float*)alloc((size_t)MBN * CC * 4);
  float*  agg       = (float*)alloc((size_t)MBN * CC * 4);
  __bf16* agg_bf    = (__bf16*)alloc((size_t)MBN * CC * 2);
  float*  nd2       = (float*)alloc((size_t)MBN * CC * 4);
  __bf16* nd2_bf    = (__bf16*)alloc((size_t)MBN * CC * 2);
  float*  gi        = (float*)alloc((size_t)MBN * 3 * CC * 4);
  float*  hG        = (float*)alloc((size_t)MRN * CC * 4);
  __bf16* hG_bf     = (__bf16*)alloc((size_t)MRN * CC * 2);
  float*  ghG       = (float*)alloc((size_t)MRN * 3 * CC * 4);
  float*  outn      = (float*)alloc((size_t)MBN * CC * 4);
  // bf16 weights, rows padded (zero-filled) to a multiple of 64 for unguarded loads
  __bf16* Wproj_bf = (__bf16*)alloc((size_t)CC * CBC * 2);
  __bf16* Wattn_bf = (__bf16*)alloc((size_t)64 * CC * 2);           // 9 -> 64 rows
  __bf16* Wgf_bf   = (__bf16*)alloc((size_t)DD * CC * 2);
  __bf16* Wih_bf   = (__bf16*)alloc((size_t)4 * DD * DD * 2);
  __bf16* Whh_bf   = (__bf16*)alloc((size_t)4 * DD * DD * 2);
  __bf16* Wkp1_bf  = (__bf16*)alloc((size_t)CC * 224 * 2);
  __bf16* Wq_bf    = (__bf16*)alloc((size_t)CC * CC * 2);
  __bf16* Wk_bf    = (__bf16*)alloc((size_t)CC * CC * 2);
  __bf16* Wv_bf    = (__bf16*)alloc((size_t)CC * CC * 2);
  __bf16* Wu_bf    = (__bf16*)alloc((size_t)CC * CC * 2);
  __bf16* Wgih_bf  = (__bf16*)alloc((size_t)3 * CC * CC * 2);
  __bf16* Wghh_bf  = (__bf16*)alloc((size_t)3 * CC * CC * 2);
  __bf16* Wback_bf = (__bf16*)alloc((size_t)CBC * CC * 2);

  // ---- host-side launch helpers ------------------------------------------
  auto tobf = [&](const float* s, __bf16* d, int rows, int cols, int sld, int dld,
                  int rows_total) {
    size_t tot = (size_t)rows_total * dld;
    int blocks = (int)((tot + 255) / 256);
    k_tobf16<<<blocks, 256, 0, stream>>>(s, d, rows, cols, sld, dld, rows_total);
  };
  auto gemm = [&](const __bf16* A, const __bf16* Bw, float* C,
                  const float* bias, const float* resid,
                  int M, int N, int Kp, int lda, int ldb, int ldc, int mode) {
    int strips = (N + 63) / 64;
    long waves = (long)(M / 16) * strips;
    int blocks = (int)((waves + 7) / 8);
    k_wmma_gemm<<<blocks, 256, 0, stream>>>(A, Bw, C, bias, resid,
                                            M, N, Kp, lda, ldb, ldc, mode);
  };
  auto fill0 = [&](float* p, size_t n) {
    int blocks = (int)((n + 255) / 256);
    k_fill0<<<blocks, 256, 0, stream>>>(p, n);
  };

  // ---- weight conversion (f32 -> bf16; K padded to %32, rows to %64) ------
  tobf(W_proj, Wproj_bf, CC, CBC, CBC, CBC, CC);
  tobf(W_attn, Wattn_bf, NN, CC, CC, CC, 64);
  tobf(W_gf,   Wgf_bf,   DD, CC, CC, CC, DD);
  tobf(W_ih,   Wih_bf,   4 * DD, DD, DD, DD, 4 * DD);
  tobf(W_hh,   Whh_bf,   4 * DD, DD, DD, DD, 4 * DD);
  tobf(W_kp1,  Wkp1_bf,  CC, HW, HW, 224, CC);
  tobf(Wq,     Wq_bf,    CC, CC, CC, CC, CC);
  tobf(Wk,     Wk_bf,    CC, CC, CC, CC, CC);
  tobf(Wv,     Wv_bf,    CC, CC, CC, CC, CC);
  tobf(Wu,     Wu_bf,    CC, CC, CC, CC, CC);
  tobf(W_gih,  Wgih_bf,  3 * CC, CC, CC, CC, 3 * CC);
  tobf(W_ghh,  Wghh_bf,  3 * CC, CC, CC, CC, 3 * CC);
  tobf(W_back, Wback_bf, CBC, CC, CC, CC, CBC);

  // ---- 1) input projection (pixel-major) ----------------------------------
  {
    size_t tot = (size_t)MPX * CBC;
    k_x_to_pxbf<<<(int)((tot + 255) / 256), 256, 0, stream>>>(x, x_bf);
  }
  gemm(x_bf, Wproj_bf, feat_px, b_proj, nullptr, MPX, CC, CBC, CBC, CBC, CC, 0);
  tobf(feat_px, feat_bf, MPX, CC, CC, CC, MPX);

  // ---- 2) attention pooling to nodes --------------------------------------
  gemm(feat_bf, Wattn_bf, logits, nullptr, nullptr, MPX, NN, CC, CC, CC, NN, 0);
  k_softmax_p<<<MBN, 256, 0, stream>>>(logits, attw);
  k_wpool<<<MBN, 256, 0, stream>>>(attw, feat_px, gf, 1.f);
  tobf(gf, gf_bf, MBN, CC, CC, CC, MBN);
  gemm(gf_bf, Wgf_bf, gfd, b_gf, nullptr, MBN, DD, CC, CC, CC, DD, 0);
  k_layernorm<<<MBN, 128, 0, stream>>>(gfd, gfd_ln, g1, beta1);

  // ---- 3) temporal LSTM ----------------------------------------------------
  tobf(gfd_ln, gfdln_bf, MBN, DD, DD, DD, MBN);
  gemm(gfdln_bf, Wih_bf, gx, b_ih, nullptr, MBN, 4 * DD, DD, DD, DD, 4 * DD, 0);
  fill0(hL, (size_t)MRN * DD);
  fill0(cL, (size_t)MRN * DD);
  for (int t = 0; t < TT; ++t) {
    tobf(hL, hL_bf, MRN, DD, DD, DD, MRN);
    gemm(hL_bf, Whh_bf, ghL, b_hh, nullptr, MRN, 4 * DD, DD, DD, DD, 4 * DD, 0);
    k_lstm_step<<<(MRN * DD + 255) / 256, 256, 0, stream>>>(gx, ghL, hL, cL, ys, t);
  }
  k_layernorm<<<MBN, 128, 0, stream>>>(ys, ys_ln, g2, beta2);

  // ---- 4) anchors / crop kernels ------------------------------------------
  k_anchors<<<MBN, 256, 0, stream>>>(ys_ln, W_pred, b_pred, alpha, khs, kws, kern2d);
  k_wpool<<<MBN, 256, 0, stream>>>(kern2d, feat_px, samp, 1.f / 9.f);
  tobf(kern2d, kern2d_bf, MBN, HW, HW, 224, MBN);
  gemm(kern2d_bf, Wkp1_bf, nodes, b_kp1, samp, MBN, CC, 224, 224, 224, CC, 1);

  // ---- 5) RSTG spatial message passing ------------------------------------
  tobf(nodes, nodes_bf, MBN, CC, CC, CC, MBN);
  gemm(nodes_bf, Wq_bf, qb, nullptr, nullptr, MBN, CC, CC, CC, CC, CC, 0);
  gemm(nodes_bf, Wk_bf, kb, nullptr, nullptr, MBN, CC, CC, CC, CC, CC, 0);
  gemm(nodes_bf, Wv_bf, vb, nullptr, nullptr, MBN, CC, CC, CC, CC, CC, 0);
  k_attn<<<128, 256, 0, stream>>>(qb, kb, vb, agg);
  tobf(agg, agg_bf, MBN, CC, CC, CC, MBN);
  gemm(agg_bf, Wu_bf, nd2, b_u, nodes, MBN, CC, CC, CC, CC, CC, 2);

  // ---- 6) temporal GRU -----------------------------------------------------
  tobf(nd2, nd2_bf, MBN, CC, CC, CC, MBN);
  gemm(nd2_bf, Wgih_bf, gi, b_gih, nullptr, MBN, 3 * CC, CC, CC, CC, 3 * CC, 0);
  fill0(hG, (size_t)MRN * CC);
  for (int t = 0; t < TT; ++t) {
    tobf(hG, hG_bf, MRN, CC, CC, CC, MRN);
    gemm(hG_bf, Wghh_bf, ghG, b_ghh, nullptr, MRN, 3 * CC, CC, CC, CC, 3 * CC, 0);
    k_gru_step<<<(MRN * CC + 255) / 256, 256, 0, stream>>>(gi, ghG, hG, outn, t);
  }

  // ---- 7) remap + back-projection + residual ------------------------------
  k_remap<<<MPX, 256, 0, stream>>>(khs, kws, outn, outmap);
  tobf(outmap, outmap_bf, MPX, CC, CC, CC, MPX);
  gemm(outmap_bf, Wback_bf, out, b_back, x, MPX, CBC, CC, CC, CC, CBC, 3);
}
